// OptimizedMoELayer_86406152061630
// MI455X (gfx1250) — matrix operand order repaired and checked by reading, not compiled, not run on hardware
//
#include <hip/hip_runtime.h>
#include <hip/hip_bf16.h>
#include <math.h>

#define HID  1024
#define FFN  4096
#define NE   8
#define MT     64            // tokens per tile (4 WMMA row-tiles -> 4x B reuse)
#define NSPLIT 4             // FFN quarters per workgroup (blockIdx.z)
#define FQ   (FFN / NSPLIT)  // 1024 FFN cols per workgroup
#define XS_LD (HID + 8)      // padded LDS strides (16B shift/row -> no bank conflicts)
#define HS_LD (FQ  + 8)

typedef __attribute__((ext_vector_type(16))) __bf16       v16bf;
typedef __attribute__((ext_vector_type(8)))  float        v8f;
typedef __attribute__((ext_vector_type(4)))  unsigned int v4u;

union Frag {                 // 32 bytes = 16 bf16 = one WMMA A/B fragment per lane
    v16bf bf;
    v4u   u[2];
};

__device__ __forceinline__ unsigned short f32_to_bf16_bits(float f) {
    unsigned int u = __float_as_uint(f);
    u += 0x7FFFu + ((u >> 16) & 1u);          // round-to-nearest-even
    return (unsigned short)(u >> 16);
}

// Branch-free exact-GELU: Abramowitz-Stegun 7.1.26 erf (max err ~2.5e-7,
// far below bf16 rounding of h). Uses v_rcp_f32 + v_exp_f32, no EXEC branches.
__device__ __forceinline__ float gelu_fast(float x) {
    float z  = x * 0.7071067811865475f;
    float az = fabsf(z);
    float t  = __builtin_amdgcn_rcpf(1.0f + 0.3275911f * az);
    float p  = t * (0.254829592f +
               t * (-0.284496736f +
               t * (1.421413741f +
               t * (-1.453152027f +
               t * 1.061405429f))));
    float er = 1.0f - p * __expf(-az * az);
    er = copysignf(er, z);
    return 0.5f * x * (1.0f + er);
}

// ---------------- x -> bf16 ----------------
__global__ void k_cvt_x(const float* __restrict__ x, unsigned short* __restrict__ xbf, int n) {
    int idx = (blockIdx.x * blockDim.x + threadIdx.x) * 4;
    if (idx + 3 < n) {
        float4 v = *(const float4*)(x + idx);
        unsigned int lo = (unsigned int)f32_to_bf16_bits(v.x) | ((unsigned int)f32_to_bf16_bits(v.y) << 16);
        unsigned int hi = (unsigned int)f32_to_bf16_bits(v.z) | ((unsigned int)f32_to_bf16_bits(v.w) << 16);
        *(uint2*)(xbf + idx) = make_uint2(lo, hi);
    }
}

// ------- W[e][K][N] f32  ->  Wt[e][N][K] bf16  (32x32 LDS tile transpose) -------
__global__ void k_cvt_transpose(const float* __restrict__ src, unsigned short* __restrict__ dst,
                                int K, int N) {
    __shared__ float tile[32][33];
    const int e = blockIdx.z;
    const size_t ebase = (size_t)e * K * N;
    const int nb = blockIdx.x * 32;
    const int kb = blockIdx.y * 32;
    for (int j = threadIdx.y; j < 32; j += 8)
        tile[j][threadIdx.x] = src[ebase + (size_t)(kb + j) * N + nb + threadIdx.x];
    __syncthreads();
    for (int j = threadIdx.y; j < 32; j += 8)
        dst[ebase + (size_t)(nb + j) * K + kb + threadIdx.x] = f32_to_bf16_bits(tile[threadIdx.x][j]);
}

// ---------------- router: one wave32 per token ----------------
__global__ void k_router(const float* __restrict__ x, const float* __restrict__ Wr,
                         int* __restrict__ counts, int* __restrict__ tok_list,
                         float* __restrict__ c_list, int T) {
    const int lane = threadIdx.x & 31;
    const int t = blockIdx.x * 8 + (threadIdx.x >> 5);
    if (t >= T) return;

    float acc[NE];
#pragma unroll
    for (int e = 0; e < NE; ++e) acc[e] = 0.f;
    const float* xr = x + (size_t)t * HID;
    for (int h = lane; h < HID; h += 32) {
        float xv = xr[h];
        const float* wr = Wr + h * NE;
#pragma unroll
        for (int e = 0; e < NE; ++e) acc[e] += xv * wr[e];
    }
#pragma unroll
    for (int off = 16; off >= 1; off >>= 1)
#pragma unroll
        for (int e = 0; e < NE; ++e) acc[e] += __shfl_xor(acc[e], off, 32);

    if (lane == 0) {
        float m = acc[0];
#pragma unroll
        for (int e = 1; e < NE; ++e) m = fmaxf(m, acc[e]);
        float p[NE];
#pragma unroll
        for (int e = 0; e < NE; ++e) p[e] = expf(acc[e] - m);
        int i0 = 0;
#pragma unroll
        for (int e = 1; e < NE; ++e) if (p[e] > p[i0]) i0 = e;   // jax top_k tie-break: lower index
        int i1 = (i0 == 0) ? 1 : 0;
#pragma unroll
        for (int e = 0; e < NE; ++e) if (e != i0 && p[e] > p[i1]) i1 = e;
        float s2 = p[i0] + p[i1];
        float c0 = p[i0] / s2, c1 = p[i1] / s2;   // softmax denom cancels in renorm
        int pos0 = atomicAdd(&counts[i0], 1);
        tok_list[i0 * T + pos0] = t; c_list[i0 * T + pos0] = c0;
        int pos1 = atomicAdd(&counts[i1], 1);
        tok_list[i1 * T + pos1] = t; c_list[i1 * T + pos1] = c1;
    }
}

// ------- fused expert FFN: async-gather -> GEMM1 -> GELU -> GEMM2 -> scatter -------
// grid = (token tiles, experts, NSPLIT FFN slices); 256 threads = 8 waves.
// Register tile per wave: 4 M-tiles x 2 N-tiles -> 8 WMMAs per (4 global + 8 ds) b128.
__global__ __launch_bounds__(256)
void k_moe_ffn(const unsigned short* __restrict__ xbf,   // [T][HID] bf16
               const unsigned short* __restrict__ w1t,   // [E][FFN][HID] bf16 (transposed)
               const float* __restrict__ b1,             // [E][FFN]
               const unsigned short* __restrict__ w2t,   // [E][HID][FFN] bf16 (transposed)
               const float* __restrict__ b2,             // [E][HID]
               const int* __restrict__ counts,
               const int* __restrict__ tok_list,
               const float* __restrict__ c_list,
               float* __restrict__ out, int T)
{
    __shared__ alignas(16) unsigned short xs[MT * XS_LD];  // ~132 KB gathered X tile (bf16)
    __shared__ alignas(16) unsigned short hs[MT * HS_LD];  // ~132 KB GELU(h) slice (bf16)
    __shared__ int   ts[MT];
    __shared__ float cs[MT];

    const int e     = blockIdx.y;
    const int split = blockIdx.z;            // FFN quarter
    const int cnt   = counts[e];
    const int base  = blockIdx.x * MT;
    if (base >= cnt) return;                 // uniform early exit

    const int tid  = threadIdx.x;
    const int lane = tid & 31;
    const int wave = tid >> 5;

    if (tid < MT) {
        int idx = base + tid;
        if (idx < cnt) { ts[tid] = tok_list[e * T + idx]; cs[tid] = c_list[e * T + idx]; }
        else           { ts[tid] = -1;                    cs[tid] = 0.f; }
    }
    __syncthreads();

    // async-stage gathered X rows into LDS (ASYNCcnt path, no VGPR round-trip)
    for (int i = tid; i < MT * (HID / 8); i += 256) {
        int r  = i >> 7;                     // HID/8 = 128 16B-chunks per row
        int ch = (i & 127) * 8;
        int t  = ts[r];
        if (t >= 0) {
            unsigned int ldsoff = (unsigned int)(uintptr_t)&xs[r * XS_LD + ch];
            const unsigned short* g = &xbf[(size_t)t * HID + ch];
            asm volatile("global_load_async_to_lds_b128 %0, %1, off"
                         :: "v"(ldsoff), "v"(g) : "memory");
        } else {
            v4u z = {0u, 0u, 0u, 0u};
            *(v4u*)&xs[r * XS_LD + ch] = z;
        }
    }
    asm volatile("s_wait_asynccnt 0x0" ::: "memory");
    __syncthreads();

    // per-lane fragment geometry (A and B both use contiguous k = 16*(lane/16) + s):
    const int arow  = lane & 15;             // A: lane -> M row
    const int koff  = (lane >> 4) << 4;      // A: K chunk 0..15 / 16..31
    const int bcol  = lane & 15;             // B: lane -> N column
    const int bko   = (lane >> 4) << 4;      // B: K chunk 0..15 / 16..31
    const int rbase = (lane >> 4) << 3;      // C/D: M = i (+8 for upper half-lanes)

    // ---- GEMM1: h[64, FQ] = gelu(X @ W1slice + b1); each wave owns FQ/8 = 128 cols ----
    for (int ntp = 0; ntp < (FQ / 8) / 32; ++ntp) {          // 4 pairs of N-tiles
        const int ncol = split * FQ + wave * (FQ / 8) + ntp * 32 + bcol; // global FFN col (tile 0)
        const int lcol = wave * (FQ / 8) + ntp * 32 + bcol;              // hs-local col (tile 0)
        v8f acc[2][4];
#pragma unroll
        for (int j = 0; j < 2; ++j)
#pragma unroll
            for (int m = 0; m < 4; ++m) acc[j][m] = (v8f){0.f,0.f,0.f,0.f,0.f,0.f,0.f,0.f};

        const unsigned short* bp = w1t + ((size_t)e * FFN + ncol) * HID + bko;
#pragma unroll 1
        for (int k0 = 0; k0 < HID; k0 += 32) {
            __builtin_prefetch(bp + k0 + 128, 0, 3);
            Frag a[4], b[2];
#pragma unroll
            for (int j = 0; j < 2; ++j) {
                b[j].u[0] = *(const v4u*)(bp + (size_t)j * 16 * HID + k0);
                b[j].u[1] = *(const v4u*)(bp + (size_t)j * 16 * HID + k0 + 8);
            }
#pragma unroll
            for (int m = 0; m < 4; ++m) {
                a[m].u[0] = *(const v4u*)&xs[(arow + 16 * m) * XS_LD + k0 + koff];
                a[m].u[1] = *(const v4u*)&xs[(arow + 16 * m) * XS_LD + k0 + koff + 8];
            }
#pragma unroll
            for (int j = 0; j < 2; ++j)
#pragma unroll
                for (int m = 0; m < 4; ++m)
                    acc[j][m] = __builtin_amdgcn_wmma_f32_16x16x32_bf16(
                        false, a[m].bf, false, b[j].bf, (short)0, acc[j][m], false, false);
        }
#pragma unroll
        for (int j = 0; j < 2; ++j) {
            float bias = b1[(size_t)e * FFN + ncol + 16 * j];
#pragma unroll
            for (int m = 0; m < 4; ++m)
#pragma unroll
                for (int i = 0; i < 8; ++i)
                    hs[(rbase + i + 16 * m) * HS_LD + lcol + 16 * j] =
                        f32_to_bf16_bits(gelu_fast(acc[j][m][i] + bias));
        }
    }
    __syncthreads();

    // ---- GEMM2 partial: y[64, HID] += h_slice @ W2slice (+ b2 on split 0), scaled scatter ----
    for (int ntp = 0; ntp < (HID / 8) / 32; ++ntp) {         // 4 pairs of N-tiles
        const int ncol = wave * (HID / 8) + ntp * 32 + bcol;             // HID col (tile 0)
        v8f acc[2][4];
#pragma unroll
        for (int j = 0; j < 2; ++j)
#pragma unroll
            for (int m = 0; m < 4; ++m) acc[j][m] = (v8f){0.f,0.f,0.f,0.f,0.f,0.f,0.f,0.f};

        const unsigned short* bp = w2t + ((size_t)e * HID + ncol) * FFN + split * FQ + bko;
#pragma unroll 1
        for (int k0 = 0; k0 < FQ; k0 += 32) {
            __builtin_prefetch(bp + k0 + 128, 0, 3);
            Frag a[4], b[2];
#pragma unroll
            for (int j = 0; j < 2; ++j) {
                b[j].u[0] = *(const v4u*)(bp + (size_t)j * 16 * FFN + k0);
                b[j].u[1] = *(const v4u*)(bp + (size_t)j * 16 * FFN + k0 + 8);
            }
#pragma unroll
            for (int m = 0; m < 4; ++m) {
                a[m].u[0] = *(const v4u*)&hs[(arow + 16 * m) * HS_LD + k0 + koff];
                a[m].u[1] = *(const v4u*)&hs[(arow + 16 * m) * HS_LD + k0 + koff + 8];
            }
#pragma unroll
            for (int j = 0; j < 2; ++j)
#pragma unroll
                for (int m = 0; m < 4; ++m)
                    acc[j][m] = __builtin_amdgcn_wmma_f32_16x16x32_bf16(
                        false, a[m].bf, false, b[j].bf, (short)0, acc[j][m], false, false);
        }
#pragma unroll
        for (int j = 0; j < 2; ++j) {
            float bias = (split == 0) ? b2[(size_t)e * HID + ncol + 16 * j] : 0.f;
#pragma unroll
            for (int m = 0; m < 4; ++m)
#pragma unroll
                for (int i = 0; i < 8; ++i) {
                    int r = rbase + i + 16 * m;
                    int t = ts[r];
                    if (t >= 0)
                        atomicAdd(&out[(size_t)t * HID + ncol + 16 * j],
                                  cs[r] * (acc[j][m][i] + bias));
                }
        }
    }
}

extern "C" void kernel_launch(void* const* d_in, const int* in_sizes, int n_in,
                              void* d_out, int out_size, void* d_ws, size_t ws_size,
                              hipStream_t stream) {
    const float* x  = (const float*)d_in[0];
    const float* Wr = (const float*)d_in[1];
    const float* W1 = (const float*)d_in[2];
    const float* b1 = (const float*)d_in[3];
    const float* W2 = (const float*)d_in[4];
    const float* b2 = (const float*)d_in[5];
    float* out = (float*)d_out;
    const int T = in_sizes[0] / HID;     // 4096

    char* ws = (char*)d_ws;
    size_t off = 0;
    unsigned short* xbf = (unsigned short*)(ws + off); off += (size_t)T * HID * 2;
    unsigned short* w1t = (unsigned short*)(ws + off); off += (size_t)NE * FFN * HID * 2;
    unsigned short* w2t = (unsigned short*)(ws + off); off += (size_t)NE * HID * FFN * 2;
    int*   counts = (int*)(ws + off);   off += 256;
    int*   tokl   = (int*)(ws + off);   off += (size_t)NE * T * sizeof(int);
    float* clist  = (float*)(ws + off); off += (size_t)NE * T * sizeof(float);

    hipMemsetAsync(out, 0, (size_t)out_size * sizeof(float), stream);
    hipMemsetAsync(counts, 0, NE * sizeof(int), stream);

    k_cvt_x<<<(T * HID / 4 + 255) / 256, 256, 0, stream>>>(x, xbf, T * HID);
    k_cvt_transpose<<<dim3(FFN / 32, HID / 32, NE), dim3(32, 8), 0, stream>>>(W1, w1t, HID, FFN);
    k_cvt_transpose<<<dim3(HID / 32, FFN / 32, NE), dim3(32, 8), 0, stream>>>(W2, w2t, FFN, HID);
    k_router<<<(T + 7) / 8, 256, 0, stream>>>(x, Wr, counts, tokl, clist, T);

    const int maxTiles = (T + MT - 1) / MT;   // worst-case skew: all tokens on one expert
    k_moe_ffn<<<dim3(maxTiles, NE, NSPLIT), 256, 0, stream>>>(xbf, w1t, b1, w2t, b2,
                                                              counts, tokl, clist, out, T);
}
